// GraphModel_14345190768783
// MI455X (gfx1250) — compile-verified, matrix operand order checked by reading, not run on hardware
//
#include <hip/hip_runtime.h>

#define NSL 5
#define CCH 256
#define HW  4096
#define TM  64
#define SA_STRIDE  520   // 512 + 8 pad (bf16) -> 1040B rows, 16B aligned, 2-way max conflicts
#define SA2_STRIDE 264   // 256 + 8 pad (bf16)
#define SU_STRIDE  260   // 256 + 4 pad (f32)

typedef __attribute__((ext_vector_type(4)))  float   f4;
typedef __attribute__((ext_vector_type(8)))  float   v8f;
typedef __attribute__((ext_vector_type(8)))  __bf16  bf8;
typedef __attribute__((ext_vector_type(16))) __bf16  v16bf;

union ABFrag { v16bf v; bf8 h[2]; };

__device__ __forceinline__ float sigm(float x) { return 1.0f / (1.0f + __expf(-x)); }
__device__ __forceinline__ float tanh_f(float x) {
  float ax = fabsf(x);
  float e  = __expf(-2.0f * ax);
  float t  = (1.0f - e) / (1.0f + e);
  return (x < 0.0f) ? -t : t;
}

// ---------------- kernel 0: weights -> bf16 ----------------
__global__ void prep_weights(const float* __restrict__ wu, const float* __restrict__ wr,
                             const float* __restrict__ wo,
                             __bf16* __restrict__ Wur, __bf16* __restrict__ Wo) {
  int idx = blockIdx.x * 256 + threadIdx.x;
  if (idx < 512 * 512) {
    int o = idx >> 9, k = idx & 511;
    float v = (o < 256) ? wu[o * 512 + k] : wr[(o - 256) * 512 + k];
    Wur[idx] = (__bf16)v;
  }
  if (idx < 256 * 512) Wo[idx] = (__bf16)wo[idx];
}

// ---------------- kernel 1: S = sum_n inputs ; D[b,n,j,hw] = <inputs[b,n], wgt_w[j]> ----------------
__global__ void pass1_sum_dots(const float* __restrict__ inp,
                               const float* __restrict__ wgt,   // [4][256]
                               float* __restrict__ S, float* __restrict__ D) {
  __shared__ float sw[4 * CCH];
  int t = threadIdx.x;
  for (int e = t; e < 4 * CCH; e += 256) sw[e] = wgt[e];
  __syncthreads();

  int b  = blockIdx.x >> 4;
  int sp = ((blockIdx.x & 15) << 8) + t;
  const float* base = inp + (size_t)b * NSL * CCH * HW + sp;

  float d[NSL][4];
#pragma unroll
  for (int n = 0; n < NSL; ++n)
#pragma unroll
    for (int j = 0; j < 4; ++j) d[n][j] = 0.0f;

  for (int c = 0; c < CCH; ++c) {
    float w0 = sw[c], w1 = sw[CCH + c], w2 = sw[2 * CCH + c], w3 = sw[3 * CCH + c];
    float ss = 0.0f;
#pragma unroll
    for (int n = 0; n < NSL; ++n) {
      float v = base[((size_t)n * CCH + c) * HW];
      ss += v;
      d[n][0] += v * w0; d[n][1] += v * w1; d[n][2] += v * w2; d[n][3] += v * w3;
    }
    S[((size_t)b * CCH + c) * HW + sp] = ss;
  }
#pragma unroll
  for (int n = 0; n < NSL; ++n)
#pragma unroll
    for (int j = 0; j < 4; ++j)
      D[(((size_t)(b * NSL + n)) * 4 + j) * HW + sp] = d[n][j];
}

// ---------------- kernel 2: fused gate + GRU via bf16 WMMA ----------------
__global__ void __launch_bounds__(256)
fused_gru(const float* __restrict__ inp, const float* __restrict__ S,
          const float* __restrict__ D,
          const __bf16* __restrict__ Wur, const __bf16* __restrict__ Wo,
          const float* __restrict__ bu, const float* __restrict__ br,
          const float* __restrict__ bo, const float* __restrict__ gammap,
          float* __restrict__ out) {
  __shared__ __align__(16) __bf16 sA [TM * SA_STRIDE];   // [m][ m_t(256) | h_t(256) ] bf16
  __shared__ __align__(16) __bf16 sA2[TM * SA2_STRIDE];  // [m][ h_t * r ] bf16
  __shared__ float sU[TM * SU_STRIDE];                   // sigmoid(u) f32
  __shared__ float sGate[TM * 4];

  const int t = threadIdx.x;
  const int i = blockIdx.y;
  const int b = blockIdx.x >> 6;
  const int spbase = (blockIdx.x & 63) * TM;

  const float* inp_i = inp + (((size_t)b * NSL + i) * CCH) * HW;
  float*       out_i = out + (((size_t)b * NSL + i) * CCH) * HW;

  // phase 0a: the 4 per-pixel gates
  {
    int m = t >> 2, j = t & 3;
    int nidx = (j < i) ? j : j + 1;
    size_t di = (((size_t)(b * NSL + i)) * 4 + j) * HW + spbase + m;
    size_t dn = (((size_t)(b * NSL + nidx)) * 4 + j) * HW + spbase + m;
    sGate[m * 4 + j] = sigm(D[di] - D[dn]);
  }
  __syncthreads();

  // phase 0b: build A = [m_t | h_t] in bf16
  {
    int q = t >> 6, m = t & 63;
    const float* Sb = S + ((size_t)b * CCH) * HW + spbase + m;
    const float* Hb = inp_i + spbase + m;
#pragma unroll 4
    for (int cc = 0; cc < 64; ++cc) {
      int c = q * 64 + cc;
      float ht = Hb[(size_t)c * HW];
      float sv = Sb[(size_t)c * HW];
      float mt = sGate[m * 4 + (c & 3)] * (sv - ht);
      sA[m * SA_STRIDE + c]       = (__bf16)mt;
      sA[m * SA_STRIDE + 256 + c] = (__bf16)ht;
    }
  }
  __syncthreads();

  const int lane  = t & 31;
  const int wv    = t >> 5;                 // wave 0..7
  const int lrow  = lane & 15;
  const int khalf = (lane & 16) ? 8 : 0;    // A/B 16-bit frag: lanes 16-31 hold K+8
  const int moff  = (lane & 16) ? 8 : 0;    // C/D: lanes 16-31 hold M+8

  // ---- GEMM1: [TM x 512] x [512 x 512] -> u (ntile<16), r (ntile>=16) ----
  for (int job = wv; job < 128; job += 8) {
    const int mtile = job & 3;
    const int ntile = job >> 2;
    v8f acc = {0.f, 0.f, 0.f, 0.f, 0.f, 0.f, 0.f, 0.f};
    const __bf16* Arow = sA  + (mtile * 16 + lrow) * SA_STRIDE;
    const __bf16* Brow = Wur + (size_t)(ntile * 16 + lrow) * 512;
#pragma unroll
    for (int kt = 0; kt < 16; ++kt) {
      int kb = kt * 32 + khalf;
      ABFrag a, bm;
      a.h[0]  = *(const bf8*)(Arow + kb);
      a.h[1]  = *(const bf8*)(Arow + kb + 16);
      bm.h[0] = *(const bf8*)(Brow + kb);
      bm.h[1] = *(const bf8*)(Brow + kb + 16);
      acc = __builtin_amdgcn_wmma_f32_16x16x32_bf16(false, a.v, false, bm.v,
                                                    (short)0, acc, false, false);
    }
    const int mg = mtile * 16 + moff;
    if (ntile < 16) {                          // u
      int c = ntile * 16 + lrow;
      float bias = bu[c];
#pragma unroll
      for (int v = 0; v < 8; ++v)
        sU[(mg + v) * SU_STRIDE + c] = sigm(acc[v] + bias);
    } else {                                   // r -> write h_t * r (bf16)
      int c = (ntile - 16) * 16 + lrow;
      float bias = br[c];
      const float* hp = inp_i + (size_t)c * HW + spbase + mg;
      f4 h0 = *(const f4*)(hp);
      f4 h1 = *(const f4*)(hp + 4);
#pragma unroll
      for (int v = 0; v < 8; ++v) {
        float r  = sigm(acc[v] + bias);
        float ht = (v < 4) ? h0[v] : h1[v - 4];
        sA2[(mg + v) * SA2_STRIDE + c] = (__bf16)(ht * r);
      }
    }
  }
  __syncthreads();

  // ---- GEMM2: [TM x 512] x [512 x 256] -> o ; final GRU combine ----
  const float gamma = gammap[0];
  for (int job = wv; job < 64; job += 8) {
    const int mtile = job & 3;
    const int ntile = job >> 2;
    v8f acc = {0.f, 0.f, 0.f, 0.f, 0.f, 0.f, 0.f, 0.f};
    const __bf16* Arow  = sA  + (mtile * 16 + lrow) * SA_STRIDE;
    const __bf16* Arow2 = sA2 + (mtile * 16 + lrow) * SA2_STRIDE;
    const __bf16* Brow  = Wo + (size_t)(ntile * 16 + lrow) * 512;
#pragma unroll
    for (int kt = 0; kt < 16; ++kt) {
      int kb = kt * 32 + khalf;
      const __bf16* ap = (kt < 8) ? (Arow + kb) : (Arow2 + (kb - 256));
      ABFrag a, bm;
      a.h[0]  = *(const bf8*)(ap);
      a.h[1]  = *(const bf8*)(ap + 16);
      bm.h[0] = *(const bf8*)(Brow + kb);
      bm.h[1] = *(const bf8*)(Brow + kb + 16);
      acc = __builtin_amdgcn_wmma_f32_16x16x32_bf16(false, a.v, false, bm.v,
                                                    (short)0, acc, false, false);
    }
    const int mg = mtile * 16 + moff;
    const int c  = ntile * 16 + lrow;
    float bias = bo[c];
    const float* hp = inp_i + (size_t)c * HW + spbase + mg;
    f4 h0 = *(const f4*)(hp);
    f4 h1 = *(const f4*)(hp + 4);
    f4 o0, o1;
#pragma unroll
    for (int v = 0; v < 8; ++v) {
      float o  = tanh_f(acc[v] + bias);
      float u  = sU[(mg + v) * SU_STRIDE + c];
      float ht = (v < 4) ? h0[v] : h1[v - 4];
      float hnew = ht * (1.0f - u) + o * u;
      float res  = hnew * gamma + ht;
      if (v < 4) o0[v] = res; else o1[v - 4] = res;
    }
    float* op = out_i + (size_t)c * HW + spbase + mg;
    *(f4*)op       = o0;
    *(f4*)(op + 4) = o1;
  }
}

extern "C" void kernel_launch(void* const* d_in, const int* in_sizes, int n_in,
                              void* d_out, int out_size, void* d_ws, size_t ws_size,
                              hipStream_t stream) {
  (void)in_sizes; (void)n_in; (void)out_size; (void)ws_size;
  const float* inp = (const float*)d_in[0];
  const float* wgt = (const float*)d_in[1];
  const float* w_r = (const float*)d_in[2];
  const float* b_r = (const float*)d_in[3];
  const float* w_u = (const float*)d_in[4];
  const float* b_u = (const float*)d_in[5];
  const float* w_o = (const float*)d_in[6];
  const float* b_o = (const float*)d_in[7];
  const float* gam = (const float*)d_in[8];
  float* out = (float*)d_out;

  char* ws = (char*)d_ws;
  float*  S   = (float*)ws;                                        // 8*256*4096*4 = 33,554,432 B
  float*  Dd  = (float*)(ws + 33554432);                           //  2,621,440 B
  __bf16* Wur = (__bf16*)(ws + 33554432 + 2621440);                //    524,288 B
  __bf16* Wo  = (__bf16*)(ws + 33554432 + 2621440 + 524288);       //    262,144 B

  prep_weights<<<1024, 256, 0, stream>>>(w_u, w_r, w_o, Wur, Wo);
  pass1_sum_dots<<<128, 256, 0, stream>>>(inp, wgt, S, Dd);
  fused_gru<<<dim3(512, 5), 256, 0, stream>>>(inp, S, Dd, Wur, Wo, b_u, b_r, b_o, gam, out);
}
